// MultiheadMaskedAttention_18657337934033
// MI455X (gfx1250) — matrix operand order repaired and checked by reading, not compiled, not run on hardware
//
#include <hip/hip_runtime.h>

// ---------------------------------------------------------------------------
// Types for CDNA5 WMMA
// ---------------------------------------------------------------------------
typedef __attribute__((ext_vector_type(16))) __bf16 v16bf;
typedef __attribute__((ext_vector_type(8)))  float  v8f;

#define HID    1024
#define NHEAD  16
#define HDIM   64
#define BATCH  2
#define SEQ    2048
#define MROWS  (BATCH * SEQ)   // 4096
#define NQKV   (3 * HID)       // 3072

// ---------------------------------------------------------------------------
// Async global->LDS staging (CDNA5 GLOBAL_LOAD_ASYNC_TO_LDS_B128, ASYNCcnt).
// The clang builtin is declared with CUDA LangAS pointer params that HIP code
// cannot spell, so issue the instruction via inline asm:
//   global_load_async_to_lds_b128 vLDS, v[addr:addr+1], off
// VDST = LDS byte offset (ptrtoint of an addrspace(3) pointer), VADDR = 64-bit
// global address. Completion is tracked by ASYNCcnt -> s_wait_asynccnt 0.
// ---------------------------------------------------------------------------
__device__ __forceinline__ void stage16B(__bf16* lds_dst, const __bf16* gsrc) {
#if defined(__gfx1250__)
    unsigned lds32 =
        (unsigned)(size_t)(__attribute__((address_space(3))) __bf16*)lds_dst;
    unsigned long long ga = (unsigned long long)(size_t)gsrc;
    asm volatile("global_load_async_to_lds_b128 %0, %1, off"
                 :: "v"(lds32), "v"(ga)
                 : "memory");
#else
    *(uint4*)lds_dst = *(const uint4*)gsrc;
#endif
}

__device__ __forceinline__ void stage_wait() {
#if defined(__gfx1250__)
    asm volatile("s_wait_asynccnt 0" ::: "memory");
#endif
}

// ---------------------------------------------------------------------------
// WMMA fragment helpers (contiguous k order: lane half h holds k0+16h..k0+16h+15 in both A and B).
// Both loaders read only CONTIGUOUS bf16 runs -> lower to ds_load_b128 pairs.
// ---------------------------------------------------------------------------
__device__ __forceinline__ v16bf load_a_frag(const __bf16* As, int lda,
                                             int lane, int k0) {
    const int o     = lane & 15;
    const int kbase = k0 + ((lane < 16) ? 0 : 16);
    const __bf16* p = As + o * lda + kbase;
    v16bf f;
#pragma unroll
    for (int i = 0; i < 8; ++i) f[i] = p[i];          // K kbase..kbase+7
#pragma unroll
    for (int i = 0; i < 8; ++i) f[8 + i] = p[8 + i];  // K kbase+8..kbase+15
    return f;
}

__device__ __forceinline__ v16bf load_b_frag_t(const __bf16* Bt, int ldb,
                                               int lane, int k0) {
    const int n     = lane & 15;
    const int kbase = k0 + ((lane < 16) ? 0 : 16);
    const __bf16* p = Bt + n * ldb + kbase;
    v16bf f;
#pragma unroll
    for (int i = 0; i < 16; ++i) f[i] = p[i];         // K kbase..kbase+15
    return f;
}

__device__ __forceinline__ v8f wmma_bf16(v16bf a, v16bf b, v8f c) {
    return __builtin_amdgcn_wmma_f32_16x16x32_bf16(
        false, a, false, b, (short)0, c, false, false);
}

// ---------------------------------------------------------------------------
// fp32 -> bf16 (plain)
// ---------------------------------------------------------------------------
__global__ void cvt_f32_bf16_kernel(const float* __restrict__ in,
                                    __bf16* __restrict__ out, int n) {
    int i = blockIdx.x * 256 + threadIdx.x;
    if (i < n) out[i] = (__bf16)in[i];
}

// fp32 [K][N] -> bf16 [N][K] transposed, LDS-tiled (32x32 tiles, 256 threads)
__global__ __launch_bounds__(256) void cvt_t_kernel(
    const float* __restrict__ in, __bf16* __restrict__ out, int K, int N) {
    __shared__ float t[32][33];
    const int nb = blockIdx.x * 32, kb = blockIdx.y * 32;
    const int tx = threadIdx.x & 31, ty = threadIdx.x >> 5;  // ty 0..7
#pragma unroll
    for (int i = 0; i < 4; ++i) {
        int k = ty + i * 8;
        t[k][tx] = in[(size_t)(kb + k) * N + nb + tx];
    }
    __syncthreads();
#pragma unroll
    for (int i = 0; i < 4; ++i) {
        int n = ty + i * 8;
        out[(size_t)(nb + n) * K + kb + tx] = (__bf16)t[tx][n];
    }
}

// ---------------------------------------------------------------------------
// QKV GEMM: [4096 x 1024] @ Wt^T + bias -> per-head bf16 Q/K (s-major), V (d-major)
// 256 threads = 8 waves, block tile 64(M) x 128(N), wave tile 32x32,
// K slab 32, double-buffered LDS with async global->LDS prefetch.
// ---------------------------------------------------------------------------
__global__ __launch_bounds__(256) void qkv_gemm_kernel(
    const __bf16* __restrict__ X,    // [4096][1024]
    const __bf16* __restrict__ Wt,   // [3072][1024] transposed weights
    const float*  __restrict__ bias, // [3072]
    __bf16* __restrict__ Q,          // [B][H][S][d]
    __bf16* __restrict__ K,          // [B][H][S][d]
    __bf16* __restrict__ V) {        // [B][H][d][S]  (transposed for attention)
    __shared__ __align__(16) __bf16 As[2][64][40];   // [buf][m][k]
    __shared__ __align__(16) __bf16 Bt[2][128][40];  // [buf][n][k]

    const int tid = threadIdx.x, lane = tid & 31, wv = tid >> 5;
    const int wr = wv & 1, wc = wv >> 1;  // 2 x 4 wave grid of 32x32 tiles
    const int rowB = blockIdx.y * 64;
    const int colB = blockIdx.x * 128;

    const int mA = tid >> 2, k8A = (tid & 3) * 8;    // A slab coords
    const int nB = tid >> 1, k16B = (tid & 1) * 16;  // B slab coords
    const __bf16* gA = &X[(size_t)(rowB + mA) * HID + k8A];
    const __bf16* gB = &Wt[(size_t)(colB + nB) * HID + k16B];

    auto stage_slab = [&](int buf, int kk) {
        stage16B(&As[buf][mA][k8A], gA + kk);
        stage16B(&Bt[buf][nB][k16B], gB + kk);
        stage16B(&Bt[buf][nB][k16B + 8], gB + kk + 8);
    };

    stage_slab(0, 0);
    stage_wait();
    __syncthreads();

    v8f acc[2][2] = {};

    for (int kk = 0; kk < HID; kk += 32) {
        const int cur = (kk >> 5) & 1;
        if (kk + 32 < HID) stage_slab(cur ^ 1, kk + 32);  // async prefetch

        v16bf a0 = load_a_frag(&As[cur][wr * 32][0],      40, lane, 0);
        v16bf a1 = load_a_frag(&As[cur][wr * 32 + 16][0], 40, lane, 0);
        v16bf b0 = load_b_frag_t(&Bt[cur][wc * 32][0],      40, lane, 0);
        v16bf b1 = load_b_frag_t(&Bt[cur][wc * 32 + 16][0], 40, lane, 0);
        acc[0][0] = wmma_bf16(a0, b0, acc[0][0]);
        acc[0][1] = wmma_bf16(a0, b1, acc[0][1]);
        acc[1][0] = wmma_bf16(a1, b0, acc[1][0]);
        acc[1][1] = wmma_bf16(a1, b1, acc[1][1]);

        stage_wait();
        __syncthreads();
    }

    // Epilogue: bias + head scatter (V stored d-major)
    const bool lo = lane < 16;
    const int  nn = lane & 15;
#pragma unroll
    for (int j = 0; j < 2; ++j) {
        const int col  = colB + wc * 32 + j * 16 + nn;  // 0..3071
        const int part = col >> 10;                     // 0=Q 1=K 2=V
        const int hcol = col & 1023;
        const int h    = hcol >> 6;
        const int d    = hcol & 63;
        const float bj = bias[col];
#pragma unroll
        for (int i = 0; i < 2; ++i) {
#pragma unroll
            for (int r = 0; r < 8; ++r) {
                int m  = rowB + wr * 32 + i * 16 + r + (lo ? 0 : 8);
                int bb = m >> 11, s = m & 2047;
                float  val = acc[i][j][r] + bj;
                size_t ho  = (size_t)(bb * NHEAD + h);
                if (part == 0)
                    Q[(ho * SEQ + s) * HDIM + d] = (__bf16)val;
                else if (part == 1)
                    K[(ho * SEQ + s) * HDIM + d] = (__bf16)val;
                else
                    V[(ho * HDIM + d) * SEQ + s] = (__bf16)val;
            }
        }
    }
}

// ---------------------------------------------------------------------------
// Flash attention: one wave per (batch, head, 32-query tile). Causal.
// K/V tiles double-buffered; next tile is async-prefetched to LDS while the
// wave runs scores + softmax + PV on the current tile.
// ---------------------------------------------------------------------------
__global__ __launch_bounds__(32) void attn_kernel(
    const __bf16* __restrict__ Q,   // [B][H][S][d]
    const __bf16* __restrict__ Kt,  // [B][H][S][d]
    const __bf16* __restrict__ Vt,  // [B][H][d][S]
    __bf16* __restrict__ O) {       // [B*S][H*d]
    __shared__ __align__(16) __bf16 Qs[32][72];      // [q][d]
    __shared__ __align__(16) __bf16 Ks[2][32][72];   // [buf][key][d]
    __shared__ __align__(16) __bf16 Vts[2][64][40];  // [buf][d][key]
    __shared__ __align__(16) float  Ss[32][33];
    __shared__ __align__(16) __bf16 Ps[32][40];
    __shared__ float row_m[32], row_l[32], row_alpha[32];

    const int lane = threadIdx.x;
    const int qt = blockIdx.x, h = blockIdx.y, b = blockIdx.z;
    const int q0 = qt * 32;

    const size_t head = (size_t)(b * NHEAD + h);
    const __bf16* Qh  = Q  + head * SEQ * HDIM;
    const __bf16* Kh  = Kt + head * SEQ * HDIM;
    const __bf16* Vh  = Vt + head * HDIM * SEQ;  // [d][s]

    auto stage_kv = [&](int buf, int k0g) {
#pragma unroll
        for (int i = 0; i < 8; ++i) {
            int e = i * 32 + lane;
            int r = e >> 3, d = (e & 7) * 8;
            stage16B(&Ks[buf][r][d], &Kh[(size_t)(k0g + r) * HDIM + d]);
            int dv = e >> 2, kq = (e & 3) * 8;
            stage16B(&Vts[buf][dv][kq], &Vh[(size_t)dv * SEQ + k0g + kq]);
        }
    };

    stage_kv(0, 0);  // async prefetch of first K/V tile overlaps Q load

    // Load 32x64 Q tile (2048 elems, 8x uint4 per lane)
#pragma unroll
    for (int i = 0; i < 8; ++i) {
        int e = i * 32 + lane, m = e >> 3, d = (e & 7) * 8;
        *(uint4*)&Qs[m][d] = *(const uint4*)&Qh[(size_t)(q0 + m) * HDIM + d];
    }
    row_m[lane] = -3.0e38f;
    row_l[lane] = 0.0f;
    stage_wait();
    __syncthreads();

    v16bf aq[2][2];
#pragma unroll
    for (int i = 0; i < 2; ++i) {
        aq[i][0] = load_a_frag(&Qs[i * 16][0], 72, lane, 0);
        aq[i][1] = load_a_frag(&Qs[i * 16][0], 72, lane, 32);
    }

    v8f o[2][4];
#pragma unroll
    for (int i = 0; i < 2; ++i)
#pragma unroll
        for (int g = 0; g < 4; ++g) o[i][g] = (v8f){};

    const bool lo = lane < 16;
    const int  nn = lane & 15;
    const int  nkt = qt + 1;  // causal: key tiles with k0 <= q0+31

    for (int kt = 0; kt < nkt; ++kt) {
        const int k0g = kt * 32;
        const int cur = kt & 1;
        if (kt + 1 < nkt) stage_kv(cur ^ 1, k0g + 32);  // async prefetch

        // Scores: 2 q-subtiles x 2 key halves, d=64 contraction (2 chained)
        v16bf bk[2][2];
#pragma unroll
        for (int kh = 0; kh < 2; ++kh) {
            bk[kh][0] = load_b_frag_t(&Ks[cur][kh * 16][0], 72, lane, 0);
            bk[kh][1] = load_b_frag_t(&Ks[cur][kh * 16][0], 72, lane, 32);
        }
#pragma unroll
        for (int i = 0; i < 2; ++i) {
#pragma unroll
            for (int kh = 0; kh < 2; ++kh) {
                v8f cs = {};
                cs = wmma_bf16(aq[i][0], bk[kh][0], cs);
                cs = wmma_bf16(aq[i][1], bk[kh][1], cs);
#pragma unroll
                for (int r = 0; r < 8; ++r) {
                    int m = i * 16 + r + (lo ? 0 : 8);
                    Ss[m][kh * 16 + nn] = cs[r] * 0.125f;  // 1/sqrt(64)
                }
            }
        }
        __syncthreads();

        // Online softmax: every lane owns one query row
        {
            const int   qg   = q0 + lane;
            const float mold = row_m[lane];
            float mnew = mold;
#pragma unroll 4
            for (int j = 0; j < 32; ++j) {
                float s = (k0g + j <= qg) ? Ss[lane][j] : -3.0e38f;
                mnew = fmaxf(mnew, s);
            }
            float lsum = 0.0f;
#pragma unroll 4
            for (int j = 0; j < 32; ++j) {
                float s = (k0g + j <= qg) ? Ss[lane][j] : -3.0e38f;
                float p = __expf(s - mnew);
                Ps[lane][j] = (__bf16)p;
                lsum += p;
            }
            float alpha      = __expf(mold - mnew);
            row_alpha[lane]  = alpha;
            row_m[lane]      = mnew;
            row_l[lane]      = row_l[lane] * alpha + lsum;
        }
        __syncthreads();

        // Rescale running O, then O += P @ V
#pragma unroll
        for (int i = 0; i < 2; ++i)
#pragma unroll
            for (int r = 0; r < 8; ++r) {
                float al = row_alpha[i * 16 + r + (lo ? 0 : 8)];
                o[i][0][r] *= al; o[i][1][r] *= al;
                o[i][2][r] *= al; o[i][3][r] *= al;
            }
        v16bf bv[4];
#pragma unroll
        for (int g = 0; g < 4; ++g)
            bv[g] = load_b_frag_t(&Vts[cur][g * 16][0], 40, lane, 0);
#pragma unroll
        for (int i = 0; i < 2; ++i) {
            v16bf ap = load_a_frag(&Ps[i * 16][0], 40, lane, 0);
#pragma unroll
            for (int g = 0; g < 4; ++g)
                o[i][g] = wmma_bf16(ap, bv[g], o[i][g]);
        }

        stage_wait();   // prefetched K/V tile complete before next iteration
        __syncthreads();
    }

    // Final normalize and store to [B*S][HID] bf16
#pragma unroll
    for (int i = 0; i < 2; ++i) {
#pragma unroll
        for (int r = 0; r < 8; ++r) {
            int   m   = i * 16 + r + (lo ? 0 : 8);
            float inv = 1.0f / row_l[m];
            size_t base = ((size_t)(b * SEQ + q0 + m)) * HID + h * HDIM;
#pragma unroll
            for (int g = 0; g < 4; ++g)
                O[base + g * 16 + nn] = (__bf16)(o[i][g][r] * inv);
        }
    }
}

// ---------------------------------------------------------------------------
// Output projection: [4096 x 1024] @ WoT^T + bias -> f32 d_out
// Same double-buffered async structure as the QKV GEMM.
// ---------------------------------------------------------------------------
__global__ __launch_bounds__(256) void oproj_gemm_kernel(
    const __bf16* __restrict__ A,    // [4096][1024] bf16 attention output
    const __bf16* __restrict__ Wt,   // [1024][1024] transposed weights
    const float*  __restrict__ bias, // [1024]
    float* __restrict__ out) {       // [4096][1024] f32
    __shared__ __align__(16) __bf16 As[2][64][40];
    __shared__ __align__(16) __bf16 Bt[2][128][40];

    const int tid = threadIdx.x, lane = tid & 31, wv = tid >> 5;
    const int wr = wv & 1, wc = wv >> 1;
    const int rowB = blockIdx.y * 64;
    const int colB = blockIdx.x * 128;

    const int mA = tid >> 2, k8A = (tid & 3) * 8;
    const int nB = tid >> 1, k16B = (tid & 1) * 16;
    const __bf16* gA = &A[(size_t)(rowB + mA) * HID + k8A];
    const __bf16* gB = &Wt[(size_t)(colB + nB) * HID + k16B];

    auto stage_slab = [&](int buf, int kk) {
        stage16B(&As[buf][mA][k8A], gA + kk);
        stage16B(&Bt[buf][nB][k16B], gB + kk);
        stage16B(&Bt[buf][nB][k16B + 8], gB + kk + 8);
    };

    stage_slab(0, 0);
    stage_wait();
    __syncthreads();

    v8f acc[2][2] = {};

    for (int kk = 0; kk < HID; kk += 32) {
        const int cur = (kk >> 5) & 1;
        if (kk + 32 < HID) stage_slab(cur ^ 1, kk + 32);

        v16bf a0 = load_a_frag(&As[cur][wr * 32][0],      40, lane, 0);
        v16bf a1 = load_a_frag(&As[cur][wr * 32 + 16][0], 40, lane, 0);
        v16bf b0 = load_b_frag_t(&Bt[cur][wc * 32][0],      40, lane, 0);
        v16bf b1 = load_b_frag_t(&Bt[cur][wc * 32 + 16][0], 40, lane, 0);
        acc[0][0] = wmma_bf16(a0, b0, acc[0][0]);
        acc[0][1] = wmma_bf16(a0, b1, acc[0][1]);
        acc[1][0] = wmma_bf16(a1, b0, acc[1][0]);
        acc[1][1] = wmma_bf16(a1, b1, acc[1][1]);

        stage_wait();
        __syncthreads();
    }

    const bool lo = lane < 16;
    const int  nn = lane & 15;
#pragma unroll
    for (int j = 0; j < 2; ++j) {
        const int col = colB + wc * 32 + j * 16 + nn;
        const float bj = bias[col];
#pragma unroll
        for (int i = 0; i < 2; ++i)
#pragma unroll
            for (int r = 0; r < 8; ++r) {
                int m = rowB + wr * 32 + i * 16 + r + (lo ? 0 : 8);
                out[(size_t)m * HID + col] = acc[i][j][r] + bj;
            }
    }
}

// ---------------------------------------------------------------------------
// Host launcher
// ---------------------------------------------------------------------------
extern "C" void kernel_launch(void* const* d_in, const int* in_sizes, int n_in,
                              void* d_out, int out_size, void* d_ws, size_t ws_size,
                              hipStream_t stream) {
    (void)in_sizes; (void)n_in; (void)out_size; (void)ws_size;
    const float* x     = (const float*)d_in[0];
    const float* w_qkv = (const float*)d_in[1];
    const float* b_qkv = (const float*)d_in[2];
    const float* w_o   = (const float*)d_in[3];
    const float* b_o   = (const float*)d_in[4];
    float* out = (float*)d_out;

    char* ws = (char*)d_ws;
    __bf16* x_bf     = (__bf16*)ws; ws += (size_t)MROWS * HID * 2;
    __bf16* wqkvT_bf = (__bf16*)ws; ws += (size_t)HID * NQKV * 2;
    __bf16* woT_bf   = (__bf16*)ws; ws += (size_t)HID * HID * 2;
    __bf16* qb       = (__bf16*)ws; ws += (size_t)MROWS * HID * 2;
    __bf16* kb       = (__bf16*)ws; ws += (size_t)MROWS * HID * 2;
    __bf16* vbT      = (__bf16*)ws; ws += (size_t)MROWS * HID * 2;
    __bf16* attn_bf  = (__bf16*)ws;

    const int n1 = MROWS * HID;
    cvt_f32_bf16_kernel<<<(n1 + 255) / 256, 256, 0, stream>>>(x, x_bf, n1);
    // weights transposed to [N][K] bf16
    cvt_t_kernel<<<dim3(NQKV / 32, HID / 32), 256, 0, stream>>>(
        w_qkv, wqkvT_bf, HID, NQKV);
    cvt_t_kernel<<<dim3(HID / 32, HID / 32), 256, 0, stream>>>(
        w_o, woT_bf, HID, HID);

    qkv_gemm_kernel<<<dim3(NQKV / 128, MROWS / 64), 256, 0, stream>>>(
        x_bf, wqkvT_bf, b_qkv, qb, kb, vbT);

    attn_kernel<<<dim3(SEQ / 32, NHEAD, BATCH), 32, 0, stream>>>(
        qb, kb, vbT, attn_bf);

    oproj_gemm_kernel<<<dim3(HID / 128, MROWS / 64), 256, 0, stream>>>(
        attn_bf, woT_bf, b_o, out);
}
